// DreamBeliefTracker_69741678953204
// MI455X (gfx1250) — matrix operand order repaired and checked by reading, not compiled, not run on hardware
//
#include <hip/hip_runtime.h>

// ---------------------------------------------------------------------------
// DreamerV3 RSSM scan for MI455X (gfx1250, wave32, WMMA bf16 16x16x32)
// Round 2: split-A concat GEMMs, single h buffer, all static inputs (obs,
// actions) converted to bf16 once in prep -> 9 launches per timestep.
// ---------------------------------------------------------------------------

#define BB      256
#define TSTEPS  64
#define STOCH   32
#define DISC    32
#define BELIEF  1024
#define LATENT  1024
#define MLPD    1024
#define ACTD    6
#define KACT    32        // action segment padded to one K-chunk
#define KIN     1056      // 1024 (z) + 32 (padded action)

typedef unsigned short ushort_t;
typedef __attribute__((ext_vector_type(16))) __bf16 v16bf;
typedef __attribute__((ext_vector_type(8)))  float  v8f;

union FragBF {
    v16bf v;
    uint4 q[2];
};

__device__ __forceinline__ ushort_t f2bf(float f) {
    unsigned u = __float_as_uint(f);
    unsigned r = (u + 0x7FFFu + ((u >> 16) & 1u)) >> 16;
    return (ushort_t)r;
}

// ---------------------------------------------------------------------------
// K-loop over one A segment. Wave owns a 32x32 patch = 2x2 WMMA tiles.
// A fragment (16-bit 16x32): lane holds row M=lane%16, 16 contiguous K
// starting at (lane>=16)*16.  B fragment (32x16): lane holds
// col N=lane%16, 16 contiguous K starting at (lane>=16)*16 (K-major weights).
// ---------------------------------------------------------------------------
__device__ __forceinline__ void gemm_kloop(
    const ushort_t* __restrict__ A, int lda,
    const ushort_t* __restrict__ Wt, int ldb,
    int Klen, int m0, int n0, int lane, v8f (&acc)[2][2])
{
    const int lrow   = lane & 15;
    const int khalfA = (lane >> 4) << 4;         // 0 or 16
    const int khalfB = (lane >> 4) << 4;         // 0 or 16

    const ushort_t* pa0 = A  + (size_t)(m0 + lrow)      * lda + khalfA;
    const ushort_t* pa1 = A  + (size_t)(m0 + 16 + lrow) * lda + khalfA;
    const ushort_t* pb0 = Wt + (size_t)(n0 + lrow)      * ldb + khalfB;
    const ushort_t* pb1 = Wt + (size_t)(n0 + 16 + lrow) * ldb + khalfB;

    for (int k0 = 0; k0 < Klen; k0 += 32) {
        FragBF a0, a1, b0, b1;
        a0.q[0] = *(const uint4*)(pa0 + k0);
        a0.q[1] = *(const uint4*)(pa0 + k0 + 8);
        a1.q[0] = *(const uint4*)(pa1 + k0);
        a1.q[1] = *(const uint4*)(pa1 + k0 + 8);
        b0.q[0] = *(const uint4*)(pb0 + k0);
        b0.q[1] = *(const uint4*)(pb0 + k0 + 8);
        b1.q[0] = *(const uint4*)(pb1 + k0);
        b1.q[1] = *(const uint4*)(pb1 + k0 + 8);

        acc[0][0] = __builtin_amdgcn_wmma_f32_16x16x32_bf16(
            false, a0.v, false, b0.v, (short)0, acc[0][0], false, false);
        acc[0][1] = __builtin_amdgcn_wmma_f32_16x16x32_bf16(
            false, a0.v, false, b1.v, (short)0, acc[0][1], false, false);
        acc[1][0] = __builtin_amdgcn_wmma_f32_16x16x32_bf16(
            false, a1.v, false, b0.v, (short)0, acc[1][0], false, false);
        acc[1][1] = __builtin_amdgcn_wmma_f32_16x16x32_bf16(
            false, a1.v, false, b1.v, (short)0, acc[1][1], false, false);
    }
}

// ---------------------------------------------------------------------------
// Split-A WMMA GEMM:
//   C[M x N] = act( [A0 | A1][M x (K0+K1)] * W^T[N x (K0+K1)] + bias )
// Workgroup tile 64(M) x 128(N); 8 waves (2M x 4N); each wave 32x32.
// K1 == 0 -> single-segment GEMM.
// ---------------------------------------------------------------------------
template <int SILU, int OUTBF>
__global__ __launch_bounds__(256)
void k_gemm(const ushort_t* __restrict__ A0, int lda0, int K0,
            const ushort_t* __restrict__ A1, int lda1, int K1,
            const ushort_t* __restrict__ Wt, int ldb,
            const float* __restrict__ bias,
            float* __restrict__ Cf, ushort_t* __restrict__ Cb, int ldc)
{
    const int lane  = threadIdx.x & 31;
    const int wave  = threadIdx.x >> 5;
    const int waveM = wave >> 2;                 // 0..1
    const int waveN = wave & 3;                  // 0..3
    const int m0    = blockIdx.x * 64 + waveM * 32;
    const int n0    = blockIdx.y * 128 + waveN * 32;

    const v8f vzero = {0.f, 0.f, 0.f, 0.f, 0.f, 0.f, 0.f, 0.f};
    v8f acc[2][2];
    acc[0][0] = vzero; acc[0][1] = vzero; acc[1][0] = vzero; acc[1][1] = vzero;

    gemm_kloop(A0, lda0, Wt, ldb, K0, m0, n0, lane, acc);
    if (K1 > 0)
        gemm_kloop(A1, lda1, Wt + K0, ldb, K1, m0, n0, lane, acc);

    const int lrow  = lane & 15;
    const int rbase = (lane >> 4) << 3;          // 0 or 8
#pragma unroll
    for (int i = 0; i < 2; ++i) {
#pragma unroll
        for (int j = 0; j < 2; ++j) {
            const int col = n0 + j * 16 + lrow;
            const float bv = bias ? bias[col] : 0.f;
#pragma unroll
            for (int v = 0; v < 8; ++v) {
                const int row = m0 + i * 16 + rbase + v;
                float x = acc[i][j][v] + bv;
                if (SILU) x = x / (1.f + __expf(-x));
                if (OUTBF) Cb[(size_t)row * ldc + col] = f2bf(x);
                else       Cf[(size_t)row * ldc + col] = x;
            }
        }
    }
}

// ---------------------------------------------------------------------------
// Weight convert + transpose: W[K x N] f32 -> W^T[N x Kpad] bf16 (zero pad)
// ---------------------------------------------------------------------------
__global__ void k_transpose(const float* __restrict__ W, ushort_t* __restrict__ Wt,
                            int K, int N, int Kpad)
{
    int idx = blockIdx.x * 256 + threadIdx.x;
    if (idx >= N * Kpad) return;
    int n = idx / Kpad;
    int k = idx - n * Kpad;
    float v = (k < K) ? W[(size_t)k * N + n] : 0.f;
    Wt[(size_t)n * Kpad + k] = f2bf(v);
}

// ---------------------------------------------------------------------------
// Bulk f32 -> bf16 convert (whole obs tensor, once per call)
// ---------------------------------------------------------------------------
__global__ void k_convert(const float* __restrict__ src, ushort_t* __restrict__ dst,
                          int n)
{
    int idx = blockIdx.x * 256 + threadIdx.x;
    if (idx < n) dst[idx] = f2bf(src[idx]);
}

// ---------------------------------------------------------------------------
// Actions -> bf16, padded 6 -> 32 per row: ACTbf[t][b][0..31]
// ---------------------------------------------------------------------------
__global__ void k_act_pack(const float* __restrict__ actions, ushort_t* __restrict__ ACTbf)
{
    int idx = blockIdx.x * 256 + threadIdx.x;
    if (idx >= TSTEPS * BB * KACT) return;
    int j  = idx & (KACT - 1);
    int tb = idx >> 5;                            // t*BB + b
    ACTbf[idx] = (j < ACTD) ? f2bf(actions[(size_t)tb * ACTD + j]) : (ushort_t)0;
}

// ---------------------------------------------------------------------------
// State init: Xin = bf16(z0); bst = b0 (f32); Hbf = bf16(b0)
// ---------------------------------------------------------------------------
__global__ void k_init(const float* __restrict__ b0, const float* __restrict__ z0,
                       ushort_t* __restrict__ Xin, ushort_t* __restrict__ Hbf,
                       float* __restrict__ bst)
{
    int idx = blockIdx.x * 256 + threadIdx.x;
    if (idx < BB * STOCH * DISC) Xin[idx] = f2bf(z0[idx]);
    if (idx < BB * BELIEF) {
        float h = b0[idx];
        bst[idx] = h;
        Hbf[idx] = f2bf(h);
    }
}

// ---------------------------------------------------------------------------
// GRU: LayerNorm(3072) over gates row, then r/c/u gate math.
// Writes h_new as f32 (state) and bf16 (single buffer feeding GRU/prior/post).
// ---------------------------------------------------------------------------
__global__ __launch_bounds__(256)
void k_gru_ln(const float* __restrict__ gates, const float* __restrict__ lns,
              const float* __restrict__ lnb, float* __restrict__ bst,
              ushort_t* __restrict__ Hbf)
{
    const int row = blockIdx.x;
    const int tid = threadIdx.x;
    const float* g = gates + (size_t)row * 3072;
    __shared__ float red[256];
    __shared__ float s_mu, s_rstd;

    float s = 0.f;
    for (int j = tid; j < 3072; j += 256) s += g[j];
    red[tid] = s; __syncthreads();
    for (int k = 128; k; k >>= 1) { if (tid < k) red[tid] += red[tid + k]; __syncthreads(); }
    if (tid == 0) s_mu = red[0] * (1.f / 3072.f);
    __syncthreads();
    const float mu = s_mu;

    float v = 0.f;
    for (int j = tid; j < 3072; j += 256) { float d = g[j] - mu; v += d * d; }
    red[tid] = v; __syncthreads();
    for (int k = 128; k; k >>= 1) { if (tid < k) red[tid] += red[tid + k]; __syncthreads(); }
    if (tid == 0) s_rstd = rsqrtf(red[0] * (1.f / 3072.f) + 1e-3f);
    __syncthreads();
    const float rstd = s_rstd;

    for (int j = tid; j < BELIEF; j += 256) {
        float r = (g[j]        - mu) * rstd * lns[j]        + lnb[j];
        float c = (g[1024 + j] - mu) * rstd * lns[1024 + j] + lnb[1024 + j];
        float u = (g[2048 + j] - mu) * rstd * lns[2048 + j] + lnb[2048 + j];
        float reset = 1.f / (1.f + __expf(-r));
        float cand  = tanhf(reset * c);
        float upd   = 1.f / (1.f + __expf(-(u - 1.f)));   // UPDATE_BIAS = -1
        float hold  = bst[(size_t)row * BELIEF + j];
        float hn    = upd * cand + (1.f - upd) * hold;
        bst[(size_t)row * BELIEF + j] = hn;
        Hbf[(size_t)row * BELIEF + j] = f2bf(hn);
    }
}

// ---------------------------------------------------------------------------
// Softmax + unimix + KL + Gumbel-max one-hot. One block per batch row,
// 32 waves; wave s = stochastic group, lane d = class.
// Forward-only: loss = (DYN+REP)*max(kl, FREE); straight-through z == one-hot.
// ---------------------------------------------------------------------------
__global__ __launch_bounds__(1024)
void k_kl_sample(const float* __restrict__ postL, const float* __restrict__ priorL,
                 const float* __restrict__ unoise, ushort_t* __restrict__ Xin,
                 float* __restrict__ out, int t)
{
    const int b = blockIdx.x;
    const int s = threadIdx.x >> 5;
    const int d = threadIdx.x & 31;
    const int idx = b * (STOCH * DISC) + s * DISC + d;

    float lp = fminf(fmaxf(postL[idx],  -20.f), 20.f);
    float lq = fminf(fmaxf(priorL[idx], -20.f), 20.f);

    // post softmax + unimix
    float m = lp;
    for (int k = 16; k; k >>= 1) m = fmaxf(m, __shfl_xor(m, k, 32));
    float e = __expf(lp - m);
    float sum = e;
    for (int k = 16; k; k >>= 1) sum += __shfl_xor(sum, k, 32);
    float post = (e / sum) * 0.99f + 0.01f / 32.f;

    // prior softmax + unimix
    m = lq;
    for (int k = 16; k; k >>= 1) m = fmaxf(m, __shfl_xor(m, k, 32));
    e = __expf(lq - m);
    sum = e;
    for (int k = 16; k; k >>= 1) sum += __shfl_xor(sum, k, 32);
    float prior = (e / sum) * 0.99f + 0.01f / 32.f;

    // KL: wave-reduce over disc, block-reduce over stoch
    float kl = post * (__logf(post + 1e-8f) - __logf(prior + 1e-8f));
    for (int k = 16; k; k >>= 1) kl += __shfl_xor(kl, k, 32);
    __shared__ float sred[STOCH];
    if (d == 0) sred[s] = kl;
    __syncthreads();
    if (threadIdx.x < 32) {
        float v = sred[threadIdx.x];
        for (int k = 16; k; k >>= 1) v += __shfl_xor(v, k, 32);
        if (threadIdx.x == 0) out[t * BB + b] = 1.1f * fmaxf(v, 0.1f);
    }

    // Gumbel-max sample -> one-hot z for next step's input GEMM
    float u = unoise[(((size_t)t * BB + b) * STOCH + s) * DISC + d];
    float g = -__logf(-__logf(u + 1e-6f) + 1e-6f);
    float score = __logf(fmaxf(post, 1e-6f)) + g;
    float best = score;
    int bi = d;
    for (int k = 16; k; k >>= 1) {
        float ob = __shfl_xor(best, k, 32);
        int   oi = __shfl_xor(bi,   k, 32);
        if (ob > best || (ob == best && oi < bi)) { best = ob; bi = oi; }
    }
    Xin[(size_t)b * (STOCH * DISC) + s * DISC + d] = f2bf(d == bi ? 1.f : 0.f);
}

// ---------------------------------------------------------------------------
// Host-side launcher
// ---------------------------------------------------------------------------
extern "C" void kernel_launch(void* const* d_in, const int* in_sizes, int n_in,
                              void* d_out, int out_size, void* d_ws, size_t ws_size,
                              hipStream_t stream)
{
    const float* b0      = (const float*)d_in[0];
    const float* z0      = (const float*)d_in[1];
    const float* actions = (const float*)d_in[2];
    const float* obs     = (const float*)d_in[3];
    const float* unoise  = (const float*)d_in[4];
    const float* Wi1 = (const float*)d_in[5];  const float* bi1 = (const float*)d_in[6];
    const float* Wi2 = (const float*)d_in[7];  const float* bi2 = (const float*)d_in[8];
    const float* Wg  = (const float*)d_in[9];
    const float* lns = (const float*)d_in[10]; const float* lnb = (const float*)d_in[11];
    const float* Wo1 = (const float*)d_in[12]; const float* bo1 = (const float*)d_in[13];
    const float* Wo2 = (const float*)d_in[14]; const float* bo2 = (const float*)d_in[15];
    const float* Wp1 = (const float*)d_in[16]; const float* bp1 = (const float*)d_in[17];
    const float* Wp2 = (const float*)d_in[18]; const float* bp2 = (const float*)d_in[19];

    char* p = (char*)d_ws;
    auto alloc = [&](size_t bytes) -> char* {
        char* r = p;
        p += (bytes + 255) & ~(size_t)255;
        return r;
    };

    // bf16 transposed weights (N x Kpad, K-major rows for WMMA B-fragments)
    ushort_t* WTi1 = (ushort_t*)alloc((size_t)1024 * KIN * 2);
    ushort_t* WTi2 = (ushort_t*)alloc((size_t)1024 * 1024 * 2);
    ushort_t* WTg  = (ushort_t*)alloc((size_t)3072 * 2048 * 2);
    ushort_t* WTo1 = (ushort_t*)alloc((size_t)1024 * 2048 * 2);
    ushort_t* WTo2 = (ushort_t*)alloc((size_t)1024 * 1024 * 2);
    ushort_t* WTp1 = (ushort_t*)alloc((size_t)1024 * 1024 * 2);
    ushort_t* WTp2 = (ushort_t*)alloc((size_t)1024 * 1024 * 2);

    // bf16 activations / state
    ushort_t* Xin   = (ushort_t*)alloc((size_t)BB * STOCH * DISC * 2);   // z (one-hot)
    ushort_t* H1    = (ushort_t*)alloc((size_t)BB * MLPD * 2);
    ushort_t* INPbf = (ushort_t*)alloc((size_t)BB * LATENT * 2);
    ushort_t* Hbf   = (ushort_t*)alloc((size_t)BB * BELIEF * 2);         // h (bf16)
    ushort_t* Hp    = (ushort_t*)alloc((size_t)BB * LATENT * 2);
    ushort_t* Ho    = (ushort_t*)alloc((size_t)BB * LATENT * 2);

    // f32 buffers
    float* gates  = (float*)alloc((size_t)BB * 3072 * 4);
    float* bst    = (float*)alloc((size_t)BB * BELIEF * 4);
    float* priorL = (float*)alloc((size_t)BB * STOCH * DISC * 4);
    float* postL  = (float*)alloc((size_t)BB * STOCH * DISC * 4);

    // bf16 static inputs, converted once per call
    ushort_t* OBSbf = (ushort_t*)alloc((size_t)TSTEPS * BB * LATENT * 2);
    ushort_t* ACTbf = (ushort_t*)alloc((size_t)TSTEPS * BB * KACT * 2);

    float* out = (float*)d_out;    // (T, B)

    // ---- prep: weight transpose/convert, static input convert, state init ----
    auto xpose = [&](const float* W, ushort_t* WT, int K, int N, int Kpad) {
        int total = N * Kpad;
        k_transpose<<<(total + 255) / 256, 256, 0, stream>>>(W, WT, K, N, Kpad);
    };
    xpose(Wi1, WTi1, 1030, 1024, KIN);
    xpose(Wi2, WTi2, 1024, 1024, 1024);
    xpose(Wg,  WTg,  2048, 3072, 2048);
    xpose(Wo1, WTo1, 2048, 1024, 2048);
    xpose(Wo2, WTo2, 1024, 1024, 1024);
    xpose(Wp1, WTp1, 1024, 1024, 1024);
    xpose(Wp2, WTp2, 1024, 1024, 1024);

    {
        int n = TSTEPS * BB * LATENT;
        k_convert<<<(n + 255) / 256, 256, 0, stream>>>(obs, OBSbf, n);
        int m = TSTEPS * BB * KACT;
        k_act_pack<<<(m + 255) / 256, 256, 0, stream>>>(actions, ACTbf);
        k_init<<<(BB * BELIEF + 255) / 256, 256, 0, stream>>>(b0, z0, Xin, Hbf, bst);
    }

    // ---- sequential scan over T (9 launches / step) ----
    const dim3 blk(256);
    for (int t = 0; t < TSTEPS; ++t) {
        const ushort_t* obs_t = OBSbf + (size_t)t * BB * LATENT;
        const ushort_t* act_t = ACTbf + (size_t)t * BB * KACT;

        // H1 = silu([z | a_t] @ Wi1 + bi1)          (256 x 1056 x 1024)
        k_gemm<1, 1><<<dim3(4, 8), blk, 0, stream>>>(
            Xin, STOCH * DISC, STOCH * DISC, act_t, KACT, KACT,
            WTi1, KIN, bi1, nullptr, H1, MLPD);

        // inp = H1 @ Wi2 + bi2                      (256 x 1024 x 1024)
        k_gemm<0, 1><<<dim3(4, 8), blk, 0, stream>>>(
            H1, MLPD, MLPD, nullptr, 0, 0,
            WTi2, 1024, bi2, nullptr, INPbf, LATENT);

        // gates = [inp | h] @ Wg  (no bias, f32)    (256 x 2048 x 3072)
        k_gemm<0, 0><<<dim3(4, 24), blk, 0, stream>>>(
            INPbf, LATENT, LATENT, Hbf, BELIEF, BELIEF,
            WTg, 2048, nullptr, gates, nullptr, 3072);

        // LayerNorm + GRU gates -> h_new (f32 state + bf16)
        k_gru_ln<<<BB, 256, 0, stream>>>(gates, lns, lnb, bst, Hbf);

        // prior: Hp = silu(h @ Wp1 + bp1); priorL = Hp @ Wp2 + bp2
        k_gemm<1, 1><<<dim3(4, 8), blk, 0, stream>>>(
            Hbf, BELIEF, BELIEF, nullptr, 0, 0,
            WTp1, 1024, bp1, nullptr, Hp, LATENT);
        k_gemm<0, 0><<<dim3(4, 8), blk, 0, stream>>>(
            Hp, LATENT, LATENT, nullptr, 0, 0,
            WTp2, 1024, bp2, priorL, nullptr, STOCH * DISC);

        // posterior: Ho = silu([obs_t | h] @ Wo1 + bo1); postL = Ho @ Wo2 + bo2
        k_gemm<1, 1><<<dim3(4, 8), blk, 0, stream>>>(
            obs_t, LATENT, LATENT, Hbf, BELIEF, BELIEF,
            WTo1, 2048, bo1, nullptr, Ho, LATENT);
        k_gemm<0, 0><<<dim3(4, 8), blk, 0, stream>>>(
            Ho, LATENT, LATENT, nullptr, 0, 0,
            WTo2, 1024, bo2, postL, nullptr, STOCH * DISC);

        // softmax/unimix/KL/loss + Gumbel-max one-hot -> next z
        k_kl_sample<<<BB, 1024, 0, stream>>>(postL, priorL, unoise, Xin, out, t);
    }
}
